// Structure_Decoder_54107997995612
// MI455X (gfx1250) — matrix operand order repaired and checked by reading, not compiled, not run on hardware
//
#include <hip/hip_runtime.h>
#include <hip/hip_bf16.h>
#include <stdint.h>

#define N_ROWS 8192
#define NHID   256
#define KSPLIT 2
#define KCH    (N_ROWS / KSPLIT)          // 4096
#define PITCH  40                          // LDS row pitch in bf16 (80B: conflict-free, 16B aligned)

typedef __attribute__((ext_vector_type(16))) __bf16 bfrag_t;
typedef __attribute__((ext_vector_type(8)))  float  v8f;
typedef __attribute__((ext_vector_type(4)))  int    v4i;

#if __has_builtin(__builtin_amdgcn_global_load_async_to_lds_b128)
#define HAS_ASYNC_LDS 1
#else
#define HAS_ASYNC_LDS 0
#endif

// ---------- bf16 helpers ----------

__device__ __forceinline__ unsigned short f2bf(float f) {
    uint32_t u = __builtin_bit_cast(uint32_t, f);
    u += 0x7FFFu + ((u >> 16) & 1u);              // round-to-nearest-even
    return (unsigned short)(u >> 16);
}

__device__ __forceinline__ uint32_t pack2_bf16(float lo, float hi) {
    uint32_t ul = __builtin_bit_cast(uint32_t, lo);
    uint32_t uh = __builtin_bit_cast(uint32_t, hi);
    ul += 0x7FFFu + ((ul >> 16) & 1u);
    uh += 0x7FFFu + ((uh >> 16) & 1u);
    return __builtin_amdgcn_perm(uh, ul, 0x07060302u);   // {uh[31:16], ul[31:16]}
}

// K index maps per CDNA5 ISA §7.12.2
// A (16x32): V0..3 -> K={0,2,4,6}, V4..7 -> K={16,18,20,22}; lanes16-31 +8
// B (32x16): same K map as A; row (M/N) = lane&15
__device__ __forceinline__ int kmapA(int v, int lane) {
    return ((v < 4) ? (2 * v) : (8 + 2 * v)) + ((lane & 16) ? 8 : 0);
}
__device__ __forceinline__ int kmapB(int v, int lane) {
    return kmapA(v, lane);
}

// raw 16-float A tile slice for one lane (two 32B chunks), for reg-pipelined cvt
struct araw_t { uint4 q[4]; };

__device__ __forceinline__ araw_t ldA_raw(const float* p0, int ld, int lane) {
    const float* p = p0 + (size_t)(lane & 15) * ld + ((lane & 16) ? 8 : 0);
    araw_t r;
    r.q[0] = *reinterpret_cast<const uint4*>(p);
    r.q[1] = *reinterpret_cast<const uint4*>(p + 4);
    r.q[2] = *reinterpret_cast<const uint4*>(p + 16);
    r.q[3] = *reinterpret_cast<const uint4*>(p + 20);
    return r;
}
__device__ __forceinline__ bfrag_t cvtA(const araw_t& r) {
    union { bfrag_t f; uint32_t w[8]; } u;
    const float* f = reinterpret_cast<const float*>(&r);
#pragma unroll
    for (int v = 0; v < 8; ++v)
        u.w[v] = pack2_bf16(f[2 * v], f[2 * v + 1]);   // matches kmapA ordering
    return u.f;
}

// fragments from an LDS panel with PITCH-bf16 rows
__device__ __forceinline__ bfrag_t ldB_lds(const unsigned short* tile, int r, int lane) {
    union { bfrag_t f; uint32_t w[8]; } u;
    const unsigned short* p = tile + r * PITCH;
#pragma unroll
    for (int v = 0; v < 8; ++v)
        u.w[v] = *reinterpret_cast<const uint32_t*>(p + kmapA(v, lane));   // same K map as A
    return u.f;
}
__device__ __forceinline__ bfrag_t ldA_lds(const unsigned short* tile, int r, int lane) {
    union { bfrag_t f; uint32_t w[8]; } u;
    const unsigned short* p = tile + r * PITCH;
#pragma unroll
    for (int v = 0; v < 8; ++v)
        u.w[v] = *reinterpret_cast<const uint32_t*>(p + kmapA(v, lane));
    return u.f;
}

// B fragment from row-major bf16 global (k contiguous)
__device__ __forceinline__ bfrag_t ldB_bf16(const unsigned short* p0, int ld, int lane) {
    union { bfrag_t f; uint32_t w[8]; } u;
    const unsigned short* p = p0 + (size_t)(lane & 15) * ld;
#pragma unroll
    for (int v = 0; v < 8; ++v)
        u.w[v] = *reinterpret_cast<const uint32_t*>(p + kmapB(v, lane));
    return u.f;
}

__device__ __forceinline__ v8f wmma_bf16(bfrag_t a, bfrag_t b, v8f c) {
    return __builtin_amdgcn_wmma_f32_16x16x32_bf16(false, a, false, b, (short)0, c,
                                                   false, false);
}

// ---------- async (or fallback) staging of one 64B row into LDS ----------
#if HAS_ASYNC_LDS
typedef __attribute__((address_space(1))) v4i* gv4i_p;   // global int4*
typedef __attribute__((address_space(3))) v4i* lv4i_p;   // LDS int4*

__device__ __forceinline__ void async_row64(const unsigned short* gsrc, unsigned short* ldst) {
    gv4i_p g = (gv4i_p)(void*)gsrc;
    lv4i_p l = (lv4i_p)(void*)ldst;
    __builtin_amdgcn_global_load_async_to_lds_b128(g, l, 0,  0);
    __builtin_amdgcn_global_load_async_to_lds_b128(g, l, 16, 0);
    __builtin_amdgcn_global_load_async_to_lds_b128(g, l, 32, 0);
    __builtin_amdgcn_global_load_async_to_lds_b128(g, l, 48, 0);
}
__device__ __forceinline__ void async_wait0() {
#if __has_builtin(__builtin_amdgcn_s_wait_asynccnt)
    __builtin_amdgcn_s_wait_asynccnt(0);
#else
    asm volatile("s_wait_asynccnt 0x0" ::: "memory");
#endif
}
#endif

__device__ __forceinline__ void sync_row64(const unsigned short* gsrc, unsigned short* ldst) {
    const uint4* s = reinterpret_cast<const uint4*>(gsrc);
    uint4* d = reinterpret_cast<uint4*>(ldst);
    d[0] = s[0]; d[1] = s[1]; d[2] = s[2]; d[3] = s[3];   // full 64B row
}

// ---------- stage 0: Wt[n][k] = bf16(W[k][n]) ----------
__global__ void k_transpose_w(const float* __restrict__ W, unsigned short* __restrict__ Wt) {
    int idx = blockIdx.x * blockDim.x + threadIdx.x;
    int n = idx >> 8, k = idx & 255;
    Wt[n * NHID + k] = f2bf(W[k * NHID + n]);
}

// ---------- stage 1: sT[n][m] = bf16((x@W)[m][n] + b[n]) ----------
__global__ void k_support(const float* __restrict__ x, const unsigned short* __restrict__ Wt,
                          const float* __restrict__ b, unsigned short* __restrict__ sT) {
    int wave = blockIdx.x * (blockDim.x >> 5) + (threadIdx.x >> 5);
    int lane = threadIdx.x & 31;
    int mBase = (wave >> 4) * 16;
    int nBase = (wave & 15) * 16;
    v8f c = {0, 0, 0, 0, 0, 0, 0, 0};
#pragma unroll
    for (int kt = 0; kt < NHID; kt += 32) {
        bfrag_t a  = cvtA(ldA_raw(x + (size_t)mBase * NHID + kt, NHID, lane));
        bfrag_t bb = ldB_bf16(Wt + (size_t)nBase * NHID + kt, NHID, lane);
        c = wmma_bf16(a, bb, c);
    }
    int n    = nBase + (lane & 15);
    int moff = (lane & 16) ? 8 : 0;
    float bias = b[n];
    unsigned short* dst = sT + (size_t)n * N_ROWS + mBase + moff;   // 8 contiguous bf16
#pragma unroll
    for (int v2 = 0; v2 < 4; ++v2)
        *reinterpret_cast<uint32_t*>(dst + 2 * v2) =
            pack2_bf16(c[2 * v2] + bias, c[2 * v2 + 1] + bias);
}

// ---------- stage 2: hPart[chunk] = adj[:, chunk] @ support[chunk, :] ----------
// Block = 8 waves = 128 adj rows, full N=256. s^T k-tiles (16 KB payload) are
// double-buffered through LDS; adj streams from HBM exactly once, with the next
// A-tile pipelined through registers while the current one feeds the WMMA pipe.
__global__ void k_aggregate(const float* __restrict__ adj, const unsigned short* __restrict__ sT,
                            float* __restrict__ hPart) {
    const int rowBlks = N_ROWS / 16 / 8;                 // 64
    int chunk  = blockIdx.x / rowBlks;                   // 0..KSPLIT-1
    int rowBlk = blockIdx.x % rowBlks;
    int waveId = threadIdx.x >> 5;
    int lane   = threadIdx.x & 31;
    int mBase  = (rowBlk * 8 + waveId) * 16;
    int k0     = chunk * KCH;

    __shared__ unsigned short ldsT[2][NHID * PITCH];     // 2 x 20 KB

    v8f acc[16];
#pragma unroll
    for (int i = 0; i < 16; ++i) acc[i] = (v8f){0, 0, 0, 0, 0, 0, 0, 0};

    const float* arow = adj + (size_t)mBase * N_ROWS + k0;

    // prologue: stage tile kt=0 into buffer 0, preload first A tile
    sync_row64(sT + (size_t)threadIdx.x * N_ROWS + k0, &ldsT[0][threadIdx.x * PITCH]);
    __syncthreads();
    araw_t cur = ldA_raw(arow, N_ROWS, lane);

    for (int kt = 0; kt < KCH; kt += 32) {
        int buf = (kt >> 5) & 1;
        int ktn = (kt + 32 < KCH) ? kt + 32 : kt;        // clamped next k-step

        // begin staging next s^T tile into the other buffer (overlapped)
        const unsigned short* nsrc = sT + (size_t)threadIdx.x * N_ROWS + k0 + ktn;
        unsigned short*       ndst = &ldsT[buf ^ 1][threadIdx.x * PITCH];
#if HAS_ASYNC_LDS
        async_row64(nsrc, ndst);
#else
        const uint4* s4 = reinterpret_cast<const uint4*>(nsrc);
        uint4 st0 = s4[0], st1 = s4[1], st2 = s4[2], st3 = s4[3];
#endif
        if (kt + 256 < KCH)
            __builtin_prefetch(arow + (size_t)(lane & 15) * N_ROWS + kt + 256, 0, 1);

        araw_t nxt = ldA_raw(arow + ktn, N_ROWS, lane);  // next A tile, in flight

        bfrag_t a = cvtA(cur);                           // convert current A tile
#pragma unroll
        for (int nt = 0; nt < 16; ++nt) {
            bfrag_t bb = ldB_lds(ldsT[buf], nt * 16 + (lane & 15), lane);
            acc[nt] = wmma_bf16(a, bb, acc[nt]);
        }

#if HAS_ASYNC_LDS
        async_wait0();
#else
        uint4* d4 = reinterpret_cast<uint4*>(ndst);
        d4[0] = st0; d4[1] = st1; d4[2] = st2; d4[3] = st3;
#endif
        cur = nxt;
        __syncthreads();                                 // next buffer ready for all waves
    }

    int moff = (lane & 16) ? 8 : 0;
    int ncol = lane & 15;
    float* dst = hPart + (size_t)chunk * N_ROWS * NHID;
#pragma unroll
    for (int nt = 0; nt < 16; ++nt)
#pragma unroll
        for (int v = 0; v < 8; ++v)
            dst[(size_t)(mBase + moff + v) * NHID + nt * 16 + ncol] = acc[nt][v];
}

// ---------- stage 2b: hB = bf16(relu(sum_k hPart[k])) ----------
__global__ void k_reduce_relu(const float* __restrict__ hPart, unsigned short* __restrict__ hB) {
    const size_t NN = (size_t)N_ROWS * NHID;
    size_t i = ((size_t)blockIdx.x * blockDim.x + threadIdx.x) * 2;
    float s0 = hPart[i]     + hPart[i + NN];
    float s1 = hPart[i + 1] + hPart[i + 1 + NN];
    *reinterpret_cast<uint32_t*>(hB + i) =
        pack2_bf16(fmaxf(s0, 0.0f), fmaxf(s1, 0.0f));
}

// ---------- stage 3: out = h @ h^T, 128x128 per block, double-buffered LDS ----------
__global__ void k_gram(const unsigned short* __restrict__ hB, float* __restrict__ out) {
    int bm = blockIdx.x & 63;
    int bn = blockIdx.x >> 6;
    int waveId = threadIdx.x >> 5;
    int lane   = threadIdx.x & 31;
    int mBase  = bm * 128 + waveId * 16;
    int nBlock = bn * 128;

    __shared__ unsigned short ldsA[2][128 * PITCH];      // 2 x 10 KB
    __shared__ unsigned short ldsB[2][128 * PITCH];      // 2 x 10 KB

    v8f acc[8];
#pragma unroll
    for (int i = 0; i < 8; ++i) acc[i] = (v8f){0, 0, 0, 0, 0, 0, 0, 0};

    int t = threadIdx.x;
    int r = t & 127;
    const unsigned short* gbase = (t < 128)
        ? hB + (size_t)(bm * 128 + r) * NHID
        : hB + (size_t)(nBlock  + r) * NHID;
    unsigned short* lbase0 = (t < 128) ? &ldsA[0][r * PITCH] : &ldsB[0][r * PITCH];
    unsigned short* lbase1 = (t < 128) ? &ldsA[1][r * PITCH] : &ldsB[1][r * PITCH];

    // prologue: stage kt=0 into buffer 0
    sync_row64(gbase, lbase0);
    __syncthreads();

    for (int kt = 0; kt < NHID; kt += 32) {
        int buf = (kt >> 5) & 1;
        int ktn = (kt + 32 < NHID) ? kt + 32 : kt;
        unsigned short* ndst = buf ? lbase0 : lbase1;
#if HAS_ASYNC_LDS
        async_row64(gbase + ktn, ndst);
#else
        const uint4* s4 = reinterpret_cast<const uint4*>(gbase + ktn);
        uint4 st0 = s4[0], st1 = s4[1], st2 = s4[2], st3 = s4[3];
#endif
        bfrag_t a = ldA_lds(ldsA[buf], waveId * 16 + (lane & 15), lane);
#pragma unroll
        for (int nt = 0; nt < 8; ++nt) {
            bfrag_t bb = ldB_lds(ldsB[buf], nt * 16 + (lane & 15), lane);
            acc[nt] = wmma_bf16(a, bb, acc[nt]);
        }
#if HAS_ASYNC_LDS
        async_wait0();
#else
        uint4* d4 = reinterpret_cast<uint4*>(ndst);
        d4[0] = st0; d4[1] = st1; d4[2] = st2; d4[3] = st3;
#endif
        __syncthreads();
    }

    int moff = (lane & 16) ? 8 : 0;
    int ncol = lane & 15;
#pragma unroll
    for (int nt = 0; nt < 8; ++nt)
#pragma unroll
        for (int v = 0; v < 8; ++v)
            out[(size_t)(mBase + moff + v) * N_ROWS + nBlock + nt * 16 + ncol] = acc[nt][v];
}

// ---------- launcher ----------
extern "C" void kernel_launch(void* const* d_in, const int* in_sizes, int n_in,
                              void* d_out, int out_size, void* d_ws, size_t ws_size,
                              hipStream_t stream) {
    const float* x   = (const float*)d_in[0];   // [8192, 256]
    const float* adj = (const float*)d_in[1];   // [8192, 8192]
    const float* W   = (const float*)d_in[2];   // [256, 256]
    const float* b   = (const float*)d_in[3];   // [256]
    float* out = (float*)d_out;                 // [8192, 8192]

    // workspace: Wt (128KB) | sT (4MB) | hB (4MB) | hPart (16MB f32)
    unsigned short* Wt = (unsigned short*)d_ws;
    unsigned short* sT = Wt + NHID * NHID;
    unsigned short* hB = sT + (size_t)NHID * N_ROWS;
    float* hPart = (float*)(hB + (size_t)N_ROWS * NHID);

    k_transpose_w<<<(NHID * NHID) / 256, 256, 0, stream>>>(W, Wt);

    k_support<<<(N_ROWS / 16) * (NHID / 16) / 8, 256, 0, stream>>>(x, Wt, b, sT);

    k_aggregate<<<(N_ROWS / 16 / 8) * KSPLIT, 256, 0, stream>>>(adj, sT, hPart);

    k_reduce_relu<<<(N_ROWS * NHID / 2) / 256, 256, 0, stream>>>(hPart, hB);

    k_gram<<<64 * 64, 256, 0, stream>>>(hB, out);
}
